// Linear_71743133712455
// MI455X (gfx1250) — matrix operand order repaired and checked by reading, not compiled, not run on hardware
//
#include <hip/hip_runtime.h>

// Problem constants (from reference)
#define NROWS 32768
#define DIN   2048
#define DOUT  2048
#define NEXP  8
#define RANK  16
#define SEG   (NROWS / NEXP)   // 4096 rows per adapter segment

// Tiling
#define BM 128
#define BN 128
#define KC 64
#define LDT 36                 // LDS row stride in uints (72 bf16 = KC + 8 pad)
#define NCHUNK (DIN / KC)      // 32 (even, so unroll-by-2 is exact)

typedef __attribute__((ext_vector_type(16))) __bf16 v16bf;
typedef __attribute__((ext_vector_type(2)))  __bf16 v2bf;
typedef __attribute__((ext_vector_type(8)))  float  v8f;

union FragU { unsigned u[8]; uint4 q[2]; v16bf v; };

#if defined(__has_builtin)
#if __has_builtin(__builtin_amdgcn_cvt_pk_bf16_f32)
#define HAVE_CVT_PK_BF16 1
#endif
#endif

// round-to-nearest-even f32 -> bf16 (scalar, epilogue only)
__device__ __forceinline__ unsigned short f2bf(float a) {
    unsigned ua = __float_as_uint(a);
    return (unsigned short)((ua + 0x7FFFu + ((ua >> 16) & 1u)) >> 16);
}
// packed f32x2 -> bf16x2: single v_cvt_pk_bf16_f32 (RNE) via builtin or asm
__device__ __forceinline__ unsigned pack2_bf16(float a, float b) {
#ifdef HAVE_CVT_PK_BF16
    union { v2bf v; unsigned u; } cv;
    cv.v = __builtin_amdgcn_cvt_pk_bf16_f32(a, b);
    return cv.u;
#else
    unsigned r;
    asm("v_cvt_pk_bf16_f32 %0, %1, %2" : "=v"(r) : "v"(a), "v"(b));
    return r;
#endif
}

// A fragment (16x32 bf16): lane L holds row M=L%16; VGPR0-3: K = (L/16)*8 + 0..7,
// VGPR4-7: K = 16 + (L/16)*8 + 0..7.  tile rows stored [row][k] contiguous bf16.
__device__ __forceinline__ v16bf load_a_frag(const unsigned* tile, int rowbase, int ks, int lane) {
    const int lrow  = lane & 15;
    const int khalf = lane >> 4;
    const unsigned* b = tile + (rowbase + lrow) * LDT;
    const int o1 = (ks >> 1) + khalf * 4;   // uints: (ks + khalf*8)/2
    FragU f;
    f.q[0] = *(const uint4*)(b + o1);
    f.q[1] = *(const uint4*)(b + o1 + 8);   // +16 bf16
    return f.v;
}

// B fragment (32x16 bf16, B[k][n] = W[n][k]): lane L holds col N=L%16;
// VGPR0-3: K = (L/16)*8 + 0..7, VGPR4-7: K = 16 + (L/16)*8 + 0..7.
__device__ __forceinline__ v16bf load_b_frag(const unsigned* tile, int rowbase, int ks, int lane) {
    const int lcol  = lane & 15;
    const int khalf = lane >> 4;
    const unsigned* b = tile + (rowbase + lcol) * LDT + (ks >> 1) + khalf * 4;
    FragU f;
    f.q[0] = *(const uint4*)(b);
    f.q[1] = *(const uint4*)(b + 8);
    return f.v;
}

__global__ __launch_bounds__(256)
void lora_linear_fused_kernel(const float* __restrict__ x,
                              const float* __restrict__ W,
                              const float* __restrict__ bias,
                              const float* __restrict__ la,   // [E,R,DIN]
                              const float* __restrict__ lb,   // [E,DOUT,R]
                              const float* __restrict__ scal, // [E]
                              float* __restrict__ out)
{
    // Double-buffered staging tiles (bf16) + epilogue buffers: ~88.5 KB LDS
    __shared__ unsigned xs_u[2][BM * LDT];
    __shared__ unsigned ws_u[2][BN * LDT];
    __shared__ unsigned as_u[2][RANK * LDT];
    __shared__ unsigned ush_u[BM * 8];      // U  (bf16 [BM][16])
    __shared__ unsigned bs_u[BN * 16];      // B_e (bf16 [BN][32], r>=16 zero)

    const int t    = threadIdx.x;
    const int lane = t & 31;
    const int wid  = t >> 5;
    const int wm   = wid >> 2;   // 0..1  (M band of 64)
    const int wn   = wid & 3;    // 0..3  (N band of 32)
    const int m0   = blockIdx.y * BM;
    const int n0   = blockIdx.x * BN;
    const int e    = m0 / SEG;               // adapter id (BM divides SEG)
    const float se = scal[e];
    const bool lead = (__builtin_amdgcn_readfirstlane(wn) == 0);  // scalar branch

    v8f acc[4][2] = {};
    v8f uacc[4]   = {};

    const int c4 = t & 15;   // float4 column within 64-wide K chunk
    const int r0 = t >> 4;   // 0..15

    // Stage one 64-wide K chunk (f32 -> bf16) into compile-time buffer bsel
    auto stage = [&](int kc, int bsel) {
        #pragma unroll
        for (int p = 0; p < 8; ++p) {
            const int row = r0 + p * 16;
            const float4 v = *(const float4*)(x + (size_t)(m0 + row) * DIN + kc + c4 * 4);
            unsigned* d = &xs_u[bsel][row * LDT + c4 * 2];
            d[0] = pack2_bf16(v.x, v.y);
            d[1] = pack2_bf16(v.z, v.w);
        }
        #pragma unroll
        for (int p = 0; p < 8; ++p) {
            const int row = r0 + p * 16;
            const float4 v = *(const float4*)(W + (size_t)(n0 + row) * DIN + kc + c4 * 4);
            unsigned* d = &ws_u[bsel][row * LDT + c4 * 2];
            d[0] = pack2_bf16(v.x, v.y);
            d[1] = pack2_bf16(v.z, v.w);
        }
        {   // scaled A_e tile: 16 x 64
            const float4 v = *(const float4*)(la + ((size_t)e * RANK + r0) * DIN + kc + c4 * 4);
            unsigned* d = &as_u[bsel][r0 * LDT + c4 * 2];
            d[0] = pack2_bf16(v.x * se, v.y * se);
            d[1] = pack2_bf16(v.z * se, v.w * se);
        }
    };

    // One 64-wide K chunk of WMMAs from buffer (compile-time pointers)
    auto compute_chunk = [&](const unsigned* xs, const unsigned* ws, const unsigned* as) {
        #pragma unroll
        for (int ks = 0; ks < KC; ks += 32) {
            v16bf a[4];
            #pragma unroll
            for (int mi = 0; mi < 4; ++mi)
                a[mi] = load_a_frag(xs, wm * 64 + mi * 16, ks, lane);
            v16bf b[2];
            #pragma unroll
            for (int ni = 0; ni < 2; ++ni)
                b[ni] = load_b_frag(ws, wn * 32 + ni * 16, ks, lane);
            #pragma unroll
            for (int mi = 0; mi < 4; ++mi)
                #pragma unroll
                for (int ni = 0; ni < 2; ++ni)
                    acc[mi][ni] = __builtin_amdgcn_wmma_f32_16x16x32_bf16(
                        false, a[mi], false, b[ni], (short)0, acc[mi][ni], false, false);
            // LoRA U = x . A_e^T  (N-dim = 16 = one WMMA tile); wn==0 waves only
            if (lead) {
                v16bf ba = load_b_frag(as, 0, ks, lane);
                #pragma unroll
                for (int mi = 0; mi < 4; ++mi)
                    uacc[mi] = __builtin_amdgcn_wmma_f32_16x16x32_bf16(
                        false, a[mi], false, ba, (short)0, uacc[mi], false, false);
            }
        }
    };

    stage(0, 0);   // prologue fill of buffer 0

    // Unrolled-by-2 ping/pong: all LDS buffer selects are compile-time constants.
    for (int i = 0; i < NCHUNK; i += 2) {
        __syncthreads();
        stage((i + 1) * KC, 1);                 // overlap HBM staging with WMMAs
        compute_chunk(xs_u[0], ws_u[0], as_u[0]);
        __syncthreads();
        if (i + 2 < NCHUNK)
            stage((i + 2) * KC, 0);
        compute_chunk(xs_u[1], ws_u[1], as_u[1]);
    }

    __syncthreads();
    // ---- publish U (f32 accum -> bf16) to LDS: C-layout row = j + (lane>=16)*8 ----
    if (lead) {
        unsigned short* ush = (unsigned short*)ush_u;
        #pragma unroll
        for (int mi = 0; mi < 4; ++mi)
            #pragma unroll
            for (int j = 0; j < 8; ++j) {
                const int row = wm * 64 + mi * 16 + j + ((lane >> 4) << 3);
                ush[row * 16 + (lane & 15)] = f2bf(uacc[mi][j]);
            }
    }
    // ---- stage B_e tile [BN][R] -> bf16, K padded 16->32 with zeros ----
    {
        const int q  = t & 3;
        const int rr = t >> 2;
        #pragma unroll
        for (int p = 0; p < 2; ++p) {
            const int row = rr + p * 64;
            const float4 v = *(const float4*)(lb + ((size_t)e * DOUT + (n0 + row)) * RANK + q * 4);
            unsigned* d = bs_u + row * 16 + q * 2;
            d[0] = pack2_bf16(v.x, v.y);
            d[1] = pack2_bf16(v.z, v.w);
            d[8] = 0u;          // zero pad r = 16..31 region
            d[9] = 0u;
        }
    }
    __syncthreads();

    // ---- delta: C += U(64x16, zero-padded K=32) x B_e^T ----
    {
        const int lrow  = lane & 15;
        const int khalf = lane >> 4;
        #pragma unroll
        for (int ni = 0; ni < 2; ++ni) {
            FragU fb;
            const unsigned* bb = bs_u + (wn * 32 + ni * 16 + lrow) * 16 + khalf * 4;
            fb.q[0] = *(const uint4*)(bb);
            fb.q[1] = *(const uint4*)(bb + 8);
            #pragma unroll
            for (int mi = 0; mi < 4; ++mi) {
                FragU fa;
                const unsigned* ab = ush_u + (wm * 64 + mi * 16 + lrow) * 8 + khalf * 4;
                fa.q[0] = *(const uint4*)(ab);
                fa.u[4] = 0u; fa.u[5] = 0u; fa.u[6] = 0u; fa.u[7] = 0u;  // K >= 16 is zero
                acc[mi][ni] = __builtin_amdgcn_wmma_f32_16x16x32_bf16(
                    false, fa.v, false, fb.v, (short)0, acc[mi][ni], false, false);
            }
        }
    }

    // ---- epilogue: + bias, store f32 ----
    #pragma unroll
    for (int ni = 0; ni < 2; ++ni) {
        const int gn = n0 + wn * 32 + ni * 16 + (lane & 15);
        const float bv = bias[gn];
        #pragma unroll
        for (int mi = 0; mi < 4; ++mi)
            #pragma unroll
            for (int j = 0; j < 8; ++j) {
                const int gm = m0 + wm * 64 + mi * 16 + j + ((lane >> 4) << 3);
                out[(size_t)gm * DOUT + gn] = acc[mi][ni][j] + bv;
            }
    }
}

extern "C" void kernel_launch(void* const* d_in, const int* in_sizes, int n_in,
                              void* d_out, int out_size, void* d_ws, size_t ws_size,
                              hipStream_t stream) {
    (void)in_sizes; (void)n_in; (void)out_size; (void)d_ws; (void)ws_size;
    const float* x    = (const float*)d_in[0];
    const float* W    = (const float*)d_in[1];
    const float* bias = (const float*)d_in[2];
    const float* la   = (const float*)d_in[3];
    const float* lb   = (const float*)d_in[4];
    const float* sc   = (const float*)d_in[5];
    float* out = (float*)d_out;

    dim3 grid(DOUT / BN, NROWS / BM);   // 16 x 256 blocks
    lora_linear_fused_kernel<<<grid, 256, 0, stream>>>(x, W, bias, la, lb, sc, out);
}
